// OperaTransformerEncoder_86655260164200
// MI455X (gfx1250) — matrix operand order repaired and checked by reading, not compiled, not run on hardware
//
#include <hip/hip_runtime.h>
#include <hip/hip_bf16.h>

// ---- problem constants (match reference) ----
#define BB   2
#define SSEQ 1024
#define FIN  768
#define DMOD 1024
#define NH   16
#define NL   6
#define DFFN 2048
#define NC   7
#define HDIM 64
#define EPSLN 1e-5f

// ---- WMMA tile config ----
#define TM 128
#define TN 128
#define TK 32
#define LSTR 40   // bf16 elements per LDS row (80B: 16B-aligned, conflict-free)

typedef __bf16 bf16_t;
typedef bf16_t v16bf __attribute__((ext_vector_type(16)));
typedef bf16_t v8bf  __attribute__((ext_vector_type(8)));
typedef float  v8f   __attribute__((ext_vector_type(8)));
typedef float  v4f   __attribute__((ext_vector_type(4)));

__device__ __forceinline__ v16bf ld_frag(const bf16_t* lo, const bf16_t* hi) {
    v8bf a = *(const v8bf*)lo;
    v8bf b = *(const v8bf*)hi;
    return __builtin_shufflevector(a, b, 0,1,2,3,4,5,6,7,8,9,10,11,12,13,14,15);
}

// =====================================================================
// Generic GEMM: C = scale * (A[M,K] @ Wt[N,K]^T) (+bias)(+residual)(relu)
// Optional causal mask (col>row -> -inf). Batched over blockIdx.z with
// separate (b, h) strides: z -> zb = z/Hz, zh = z%Hz.
// Double-buffered LDS pipeline: global loads for slab k+1 stay in flight
// while the WMMAs for slab k execute; one barrier per K-step.
// All of M%128, N%128, K%32 == 0 for every call in this model.
// =====================================================================
__global__ void __launch_bounds__(256)
gemm_wmma_bf16(const float* __restrict__ A, long aOffB, long aOffH, int lda,
               const float* __restrict__ Wt, long wOffB, long wOffH, int ldw,
               const float* __restrict__ bias,
               const float* __restrict__ residual, int ldr,
               float* __restrict__ Cout, long cOffB, long cOffH, int ldc,
               int M, int N, int K, int Hz,
               float scale, int relu, int causal)
{
    __shared__ __align__(16) bf16_t As[2][TM * LSTR];
    __shared__ __align__(16) bf16_t Bs[2][TN * LSTR];

    const int z  = blockIdx.z;
    const int zb = z / Hz, zh = z % Hz;
    A    += aOffB * zb + aOffH * zh;
    Wt   += wOffB * zb + wOffH * zh;
    Cout += cOffB * zb + cOffH * zh;

    const int tid  = threadIdx.x;
    const int wave = tid >> 5;
    const int lane = tid & 31;
    const int half = lane >> 4;   // 0 or 1
    const int l16  = lane & 15;
    const int wm   = (wave & 3) * 32;   // wave M offset in tile
    const int wn   = (wave >> 2) * 64;  // wave N offset in tile
    const int m0   = blockIdx.y * TM;
    const int n0   = blockIdx.x * TN;

    // staging geometry: thread owns column scol, rows srow + 32*c
    const int srow = tid >> 3;        // 0..31
    const int scol = (tid & 7) * 4;   // 0..28
    const float* aPtr = A  + (size_t)(m0 + srow) * lda + scol;
    const float* wPtr = Wt + (size_t)(n0 + srow) * ldw + scol;
    const size_t aStep = (size_t)32 * lda;
    const size_t wStep = (size_t)32 * ldw;

    v8f acc[2][4];
#pragma unroll
    for (int i = 0; i < 2; ++i)
#pragma unroll
        for (int j = 0; j < 4; ++j)
#pragma unroll
            for (int e = 0; e < 8; ++e) acc[i][j][e] = 0.0f;

    v4f ra[4], rb[4];
    auto issue_loads = [&]() {
#pragma unroll
        for (int c = 0; c < 4; ++c) ra[c] = *(const v4f*)(aPtr + c * aStep);
#pragma unroll
        for (int c = 0; c < 4; ++c) rb[c] = *(const v4f*)(wPtr + c * wStep);
        aPtr += TK;
        wPtr += TK;
    };
    auto stage = [&](int buf) {
#pragma unroll
        for (int c = 0; c < 4; ++c) {
            bf16_t* d = &As[buf][(srow + 32 * c) * LSTR + scol];
            d[0] = (bf16_t)ra[c][0]; d[1] = (bf16_t)ra[c][1];
            d[2] = (bf16_t)ra[c][2]; d[3] = (bf16_t)ra[c][3];
        }
#pragma unroll
        for (int c = 0; c < 4; ++c) {
            bf16_t* d = &Bs[buf][(srow + 32 * c) * LSTR + scol];
            d[0] = (bf16_t)rb[c][0]; d[1] = (bf16_t)rb[c][1];
            d[2] = (bf16_t)rb[c][2]; d[3] = (bf16_t)rb[c][3];
        }
    };

    const int nk = K / TK;
    issue_loads();
    stage(0);
    __syncthreads();

    for (int ks = 0; ks < nk; ++ks) {
        const int  cur  = ks & 1;
        const bool more = (ks + 1 < nk);
        if (more) {
            issue_loads();   // slab ks+1 in flight across the WMMAs below
            __builtin_prefetch(aPtr, 0, 1);   // gfx1250 global_prefetch_b8
            __builtin_prefetch(wPtr, 0, 1);
        }

        // A fragments: lane l16 = M row; elems 0..7 -> K=half*8+e, 8..15 -> +16
        v16bf af[2];
#pragma unroll
        for (int mi = 0; mi < 2; ++mi) {
            const bf16_t* base = &As[cur][(wm + mi * 16 + l16) * LSTR];
            af[mi] = ld_frag(base + half * 8, base + 16 + half * 8);
        }
#pragma unroll
        for (int ni = 0; ni < 4; ++ni) {
            // B fragment: lane l16 = N col; elems 0..7 -> K=half*8+e, 8..15 -> +16
            const bf16_t* base = &Bs[cur][(wn + ni * 16 + l16) * LSTR];
            v16bf bfr = ld_frag(base + half * 8, base + 16 + half * 8);
#pragma unroll
            for (int mi = 0; mi < 2; ++mi)
                acc[mi][ni] = __builtin_amdgcn_wmma_f32_16x16x32_bf16(
                    false, af[mi], false, bfr, (short)0, acc[mi][ni], false, false);
        }

        if (more) stage(cur ^ 1);   // waits for loads only here, after MMAs
        __syncthreads();
    }

    // epilogue: C/D layout -> VGPR r holds M = r + 8*half, lane l16 = N
#pragma unroll
    for (int mi = 0; mi < 2; ++mi)
#pragma unroll
        for (int ni = 0; ni < 4; ++ni)
#pragma unroll
            for (int r = 0; r < 8; ++r) {
                int row = m0 + wm + mi * 16 + half * 8 + r;
                int col = n0 + wn + ni * 16 + l16;
                float v = acc[mi][ni][r] * scale;
                if (bias)     v += bias[col];
                if (residual) v += residual[(size_t)row * ldr + col];
                if (relu)     v = fmaxf(v, 0.0f);
                if (causal && col > row) v = -__builtin_huge_valf();
                Cout[(size_t)row * ldc + col] = v;
            }
}

// =====================================================================
// O = P @ V per (b,h):  P = attnbuf[b,h] (S x S f32),
// V rows j live at qkv[b, j, 2D + h*64 .. +64].  N = 64 (head dim).
// V tile is transposed into LDS so B-fragments are contiguous.
// Same double-buffered pipeline as the GEMM.
// =====================================================================
__global__ void __launch_bounds__(256)
attn_v_wmma(const float* __restrict__ attnbuf, const float* __restrict__ qkv,
            float* __restrict__ obuf)
{
    __shared__ __align__(16) bf16_t As[2][TM * LSTR];   // 128 x 32 of P
    __shared__ __align__(16) bf16_t Bs[2][64 * LSTR];   // 64(n) x 32(k) = V^T

    const int z = blockIdx.y;
    const int b = z / NH, hh = z % NH;
    const float* P = attnbuf + (size_t)z * SSEQ * SSEQ;
    const float* V = qkv + (size_t)b * SSEQ * 3 * DMOD + 2 * DMOD + hh * HDIM;
    float*       O = obuf + (size_t)b * SSEQ * DMOD + hh * HDIM;
    const int m0 = blockIdx.x * TM;

    const int tid  = threadIdx.x;
    const int wave = tid >> 5;
    const int lane = tid & 31;
    const int half = lane >> 4;
    const int l16  = lane & 15;

    // P staging: thread owns column scol, rows srow + 32*c
    const int srow = tid >> 3;
    const int scol = (tid & 7) * 4;
    const float* pPtr = P + (size_t)(m0 + srow) * SSEQ + scol;
    // V staging: thread owns rows kr0 + 16*c, cols nc..nc+3 (stored transposed)
    const int kr0 = tid >> 4;          // 0..15
    const int nc  = (tid & 15) * 4;    // 0..60
    const float* vPtr = V + (size_t)kr0 * (3 * DMOD) + nc;

    v8f acc[4];
#pragma unroll
    for (int j = 0; j < 4; ++j)
#pragma unroll
        for (int e = 0; e < 8; ++e) acc[j][e] = 0.0f;

    v4f rp[4], rv[2];
    auto issue_loads = [&]() {
#pragma unroll
        for (int c = 0; c < 4; ++c) rp[c] = *(const v4f*)(pPtr + (size_t)(32 * c) * SSEQ);
#pragma unroll
        for (int c = 0; c < 2; ++c) rv[c] = *(const v4f*)(vPtr + (size_t)(16 * c) * (3 * DMOD));
        pPtr += TK;
        vPtr += (size_t)TK * (3 * DMOD);
    };
    auto stage = [&](int buf) {
#pragma unroll
        for (int c = 0; c < 4; ++c) {
            bf16_t* d = &As[buf][(srow + 32 * c) * LSTR + scol];
            d[0] = (bf16_t)rp[c][0]; d[1] = (bf16_t)rp[c][1];
            d[2] = (bf16_t)rp[c][2]; d[3] = (bf16_t)rp[c][3];
        }
#pragma unroll
        for (int c = 0; c < 2; ++c)
#pragma unroll
            for (int j = 0; j < 4; ++j)
                Bs[buf][(nc + j) * LSTR + (kr0 + 16 * c)] = (bf16_t)rv[c][j];
    };

    const int nk = SSEQ / TK;
    issue_loads();
    stage(0);
    __syncthreads();

    for (int ks = 0; ks < nk; ++ks) {
        const int  cur  = ks & 1;
        const bool more = (ks + 1 < nk);
        if (more) issue_loads();

        const bf16_t* abase = &As[cur][(wave * 16 + l16) * LSTR];
        v16bf af = ld_frag(abase + half * 8, abase + 16 + half * 8);
#pragma unroll
        for (int ni = 0; ni < 4; ++ni) {
            const bf16_t* bbase = &Bs[cur][(ni * 16 + l16) * LSTR];
            v16bf bfr = ld_frag(bbase + half * 8, bbase + 16 + half * 8);
            acc[ni] = __builtin_amdgcn_wmma_f32_16x16x32_bf16(
                false, af, false, bfr, (short)0, acc[ni], false, false);
        }

        if (more) stage(cur ^ 1);
        __syncthreads();
    }

#pragma unroll
    for (int ni = 0; ni < 4; ++ni)
#pragma unroll
        for (int r = 0; r < 8; ++r) {
            int row = m0 + wave * 16 + half * 8 + r;
            int col = ni * 16 + l16;
            O[(size_t)row * DMOD + col] = acc[ni][r];
        }
}

// =====================================================================
// Row softmax over attnbuf rows (masked entries hold -inf -> exp = 0)
// =====================================================================
__global__ void __launch_bounds__(256)
softmax_rows(float* __restrict__ attnbuf)
{
    const int i = blockIdx.x;       // row
    const int z = blockIdx.y;       // b*NH + h
    float* row = attnbuf + ((size_t)z * SSEQ + i) * SSEQ;
    __shared__ float sred[256];
    const int tid = threadIdx.x;

    float m = -__builtin_huge_valf();
    for (int j = tid; j < SSEQ; j += 256) m = fmaxf(m, row[j]);
    sred[tid] = m; __syncthreads();
    for (int st = 128; st > 0; st >>= 1) {
        if (tid < st) sred[tid] = fmaxf(sred[tid], sred[tid + st]);
        __syncthreads();
    }
    m = sred[0]; __syncthreads();

    float sum = 0.0f;
    for (int j = tid; j < SSEQ; j += 256) {
        float e = __expf(row[j] - m);
        row[j] = e;
        sum += e;
    }
    sred[tid] = sum; __syncthreads();
    for (int st = 128; st > 0; st >>= 1) {
        if (tid < st) sred[tid] += sred[tid + st];
        __syncthreads();
    }
    const float inv = 1.0f / sred[0];
    for (int j = tid; j < SSEQ; j += 256) row[j] *= inv;
}

// avg over heads: out[b,i,j] = mean_h attnbuf[b,h,i,j]
__global__ void avg_heads(const float* __restrict__ attnbuf, float* __restrict__ out)
{
    size_t idx = (size_t)blockIdx.x * blockDim.x + threadIdx.x;
    const size_t total = (size_t)BB * SSEQ * SSEQ;
    if (idx >= total) return;
    size_t b   = idx / ((size_t)SSEQ * SSEQ);
    size_t rem = idx % ((size_t)SSEQ * SSEQ);
    float s = 0.0f;
    for (int hh = 0; hh < NH; ++hh)
        s += attnbuf[(b * NH + hh) * (size_t)SSEQ * SSEQ + rem];
    out[idx] = s * (1.0f / NH);
}

// LayerNorm over last dim D
__global__ void __launch_bounds__(256)
layernorm_k(const float* __restrict__ y, const float* __restrict__ g,
            const float* __restrict__ bta, float* __restrict__ out)
{
    const int rowi = blockIdx.x;
    const float* yr = y + (size_t)rowi * DMOD;
    float* orow = out + (size_t)rowi * DMOD;
    __shared__ float sred[256];
    const int tid = threadIdx.x;

    float s = 0.0f;
    for (int j = tid; j < DMOD; j += 256) s += yr[j];
    sred[tid] = s; __syncthreads();
    for (int st = 128; st > 0; st >>= 1) {
        if (tid < st) sred[tid] += sred[tid + st];
        __syncthreads();
    }
    const float mu = sred[0] * (1.0f / DMOD);
    __syncthreads();

    float v = 0.0f;
    for (int j = tid; j < DMOD; j += 256) { float d = yr[j] - mu; v += d * d; }
    sred[tid] = v; __syncthreads();
    for (int st = 128; st > 0; st >>= 1) {
        if (tid < st) sred[tid] += sred[tid + st];
        __syncthreads();
    }
    const float inv = rsqrtf(sred[0] * (1.0f / DMOD) + EPSLN);
    for (int j = tid; j < DMOD; j += 256)
        orow[j] = (yr[j] - mu) * inv * g[j] + bta[j];
}

// tiny classifier heads (C = 7): out[row, c] = h[row,:] . Wc[c,:] + bc[c]
__global__ void __launch_bounds__(256)
head_k(const float* __restrict__ h, int ldh, const float* __restrict__ Wc,
       const float* __restrict__ bc, float* __restrict__ out, int Kdim)
{
    const int rowi = blockIdx.x;
    const float* hr = h + (size_t)rowi * ldh;
    __shared__ float sred[256];
    const int tid = threadIdx.x;
    for (int c = 0; c < NC; ++c) {
        float s = 0.0f;
        for (int j = tid; j < Kdim; j += 256) s += hr[j] * Wc[(size_t)c * Kdim + j];
        sred[tid] = s; __syncthreads();
        for (int st = 128; st > 0; st >>= 1) {
            if (tid < st) sred[tid] += sred[tid + st];
            __syncthreads();
        }
        if (tid == 0) out[(size_t)rowi * NC + c] = sred[0] + bc[c];
        __syncthreads();
    }
}

// =====================================================================
extern "C" void kernel_launch(void* const* d_in, const int* in_sizes, int n_in,
                              void* d_out, int out_size, void* d_ws, size_t ws_size,
                              hipStream_t stream)
{
    const float* x     = (const float*)d_in[0];
    const float* in_W  = (const float*)d_in[1];
    const float* in_b  = (const float*)d_in[2];
    const float* Wqkv  = (const float*)d_in[3];
    const float* bqkv  = (const float*)d_in[4];
    const float* Wo    = (const float*)d_in[5];
    const float* bo    = (const float*)d_in[6];
    const float* ln1g  = (const float*)d_in[7];
    const float* ln1b  = (const float*)d_in[8];
    const float* W1    = (const float*)d_in[9];
    const float* b1    = (const float*)d_in[10];
    const float* W2    = (const float*)d_in[11];
    const float* b2    = (const float*)d_in[12];
    const float* ln2g  = (const float*)d_in[13];
    const float* ln2b  = (const float*)d_in[14];
    const float* clsW  = (const float*)d_in[15];
    const float* clsb  = (const float*)d_in[16];
    const float* featW = (const float*)d_in[17];
    const float* featb = (const float*)d_in[18];

    float* out      = (float*)d_out;
    float* logits   = out;
    float* fe       = out + (size_t)BB * SSEQ * NC;
    float* attn_out = fe  + (size_t)BB * SSEQ * NC;

    float* ws    = (float*)d_ws;
    float* hbuf  = ws;                                   // B*S*D
    float* qkvb  = hbuf + (size_t)BB * SSEQ * DMOD;      // B*S*3D
    float* obuf  = qkvb + (size_t)BB * SSEQ * 3 * DMOD;  // B*S*D
    float* ffb   = obuf + (size_t)BB * SSEQ * DMOD;      // B*S*DFF
    float* ybuf  = ffb  + (size_t)BB * SSEQ * DFFN;      // B*S*D
    float* attnb = ybuf + (size_t)BB * SSEQ * DMOD;      // B*H*S*S

    const dim3 blk(256);
    const int Mrows = BB * SSEQ;  // 2048

    // fe_logits = x @ feat_W^T + feat_b
    head_k<<<Mrows, blk, 0, stream>>>(x, FIN, featW, featb, fe, FIN);

    // h = x @ in_W^T + in_b
    gemm_wmma_bf16<<<dim3(DMOD / 128, Mrows / 128, 1), blk, 0, stream>>>(
        x, 0, 0, FIN, in_W, 0, 0, FIN, in_b, nullptr, 0,
        hbuf, 0, 0, DMOD, Mrows, DMOD, FIN, 1, 1.0f, 0, 0);

    for (int l = 0; l < NL; ++l) {
        const float* Wqkv_l = Wqkv + (size_t)l * 3 * DMOD * DMOD;
        const float* bqkv_l = bqkv + (size_t)l * 3 * DMOD;
        const float* Wo_l   = Wo   + (size_t)l * DMOD * DMOD;
        const float* bo_l   = bo   + (size_t)l * DMOD;
        const float* W1_l   = W1   + (size_t)l * DFFN * DMOD;
        const float* b1_l   = b1   + (size_t)l * DFFN;
        const float* W2_l   = W2   + (size_t)l * DMOD * DFFN;
        const float* b2_l   = b2   + (size_t)l * DMOD;

        // qkv = h @ Wqkv^T + bqkv
        gemm_wmma_bf16<<<dim3(3 * DMOD / 128, Mrows / 128, 1), blk, 0, stream>>>(
            hbuf, 0, 0, DMOD, Wqkv_l, 0, 0, DMOD, bqkv_l, nullptr, 0,
            qkvb, 0, 0, 3 * DMOD, Mrows, 3 * DMOD, DMOD, 1, 1.0f, 0, 0);

        // scores[b,h] = (Q K^T)/8 with causal mask, batched over z = b*NH + h
        gemm_wmma_bf16<<<dim3(SSEQ / 128, SSEQ / 128, BB * NH), blk, 0, stream>>>(
            qkvb,        (long)SSEQ * 3 * DMOD, (long)HDIM, 3 * DMOD,
            qkvb + DMOD, (long)SSEQ * 3 * DMOD, (long)HDIM, 3 * DMOD,
            nullptr, nullptr, 0,
            attnb, (long)NH * SSEQ * SSEQ, (long)SSEQ * SSEQ, SSEQ,
            SSEQ, SSEQ, HDIM, NH, 0.125f, 0, 1);

        softmax_rows<<<dim3(SSEQ, BB * NH), blk, 0, stream>>>(attnb);

        {
            size_t tot = (size_t)BB * SSEQ * SSEQ;
            avg_heads<<<(unsigned)((tot + 255) / 256), blk, 0, stream>>>(
                attnb, attn_out + (size_t)l * BB * SSEQ * SSEQ);
        }

        attn_v_wmma<<<dim3(SSEQ / 128, BB * NH), blk, 0, stream>>>(attnb, qkvb, obuf);

        // y = o @ Wo^T + bo + h ; h = LN1(y)
        gemm_wmma_bf16<<<dim3(DMOD / 128, Mrows / 128, 1), blk, 0, stream>>>(
            obuf, 0, 0, DMOD, Wo_l, 0, 0, DMOD, bo_l, hbuf, DMOD,
            ybuf, 0, 0, DMOD, Mrows, DMOD, DMOD, 1, 1.0f, 0, 0);
        layernorm_k<<<Mrows, blk, 0, stream>>>(ybuf, ln1g + (size_t)l * DMOD,
                                               ln1b + (size_t)l * DMOD, hbuf);

        // ff = relu(h @ W1^T + b1)
        gemm_wmma_bf16<<<dim3(DFFN / 128, Mrows / 128, 1), blk, 0, stream>>>(
            hbuf, 0, 0, DMOD, W1_l, 0, 0, DMOD, b1_l, nullptr, 0,
            ffb, 0, 0, DFFN, Mrows, DFFN, DMOD, 1, 1.0f, 1, 0);

        // y = ff @ W2^T + b2 + h ; h = LN2(y)
        gemm_wmma_bf16<<<dim3(DMOD / 128, Mrows / 128, 1), blk, 0, stream>>>(
            ffb, 0, 0, DFFN, W2_l, 0, 0, DFFN, b2_l, hbuf, DMOD,
            ybuf, 0, 0, DMOD, Mrows, DMOD, DFFN, 1, 1.0f, 0, 0);
        layernorm_k<<<Mrows, blk, 0, stream>>>(ybuf, ln2g + (size_t)l * DMOD,
                                               ln2b + (size_t)l * DMOD, hbuf);
    }

    // logits = h @ cls_W^T + cls_b
    head_k<<<Mrows, blk, 0, stream>>>(hbuf, DMOD, clsW, clsb, logits, DMOD);
}
